// BaseGenerator_71451075936296
// MI455X (gfx1250) — matrix operand order repaired and checked by reading, not compiled, not run on hardware
//
#include <hip/hip_runtime.h>
#include <hip/hip_bf16.h>
#include <math.h>

// ---------------------------------------------------------------------------
// Types
// ---------------------------------------------------------------------------
typedef _Float16 f16;
typedef __attribute__((ext_vector_type(16))) _Float16 v16h;
typedef __attribute__((ext_vector_type(8)))  float    v8f;
typedef __attribute__((ext_vector_type(4)))  unsigned int u32x4;
typedef __attribute__((ext_vector_type(8)))  int      i32x8;
typedef __attribute__((ext_vector_type(4)))  int      i32x4;

union FragU { v16h v; u32x4 q[2]; };
union H8    { u32x4 q; f16 h[8]; };

__device__ inline v8f zero8() {
    v8f z = {0.f, 0.f, 0.f, 0.f, 0.f, 0.f, 0.f, 0.f};
    return z;
}

// A fragment (16x32, M x K): lane = M row; element e -> K = e + (e>=8 ? 8:0) + 8*hi
__device__ inline v16h ldfrag_a(const f16* base, int stride, int lane) {
    const int r = lane & 15, hi = lane >> 4;
    const f16* p = base + (size_t)r * stride + 8 * hi;
    FragU u;
    u.q[0] = *(const u32x4*)(p);        // K = 8*hi + 0..7
    u.q[1] = *(const u32x4*)(p + 16);   // K = 8*hi + 16..23
    return u.v;
}

// B fragment (32x16, K x N), source rows are N-rows with contiguous K:
// lane = N col; element e -> K = e + (e>=8 ? 8:0) + 8*hi
__device__ inline v16h ldfrag_b(const f16* base, int stride, int lane) {
    const int n = lane & 15, hi = lane >> 4;
    const f16* p = base + (size_t)n * stride + 8 * hi;
    FragU u;
    u.q[0] = *(const u32x4*)(p);
    u.q[1] = *(const u32x4*)(p + 16);
    return u.v;
}

__device__ inline v8f wmma_f16(v16h a, v16h b, v8f c) {
    return __builtin_amdgcn_wmma_f32_16x16x32_f16(
        /*neg_a=*/false, a, /*neg_b=*/false, b,
        /*c_mod=*/(short)0, c, /*reuse_a=*/false, /*reuse_b=*/false);
}

// ---------------------------------------------------------------------------
// Tensor Data Mover: 2D f16 tile load Global -> LDS (ISA 08_async_tensor §8).
// Descriptor built per D# group layouts; pad fields reproduce the 32->40 half
// LDS row padding (16 DWORDs data, 4 DWORDs pad). OOB rows/cols read as zero.
// ---------------------------------------------------------------------------
#if __has_builtin(__builtin_amdgcn_tensor_load_to_lds) && \
    __has_builtin(__builtin_amdgcn_s_wait_tensorcnt)
#define HAVE_TDM 1
#else
#define HAVE_TDM 0
#endif

#if HAVE_TDM
__device__ inline void tdm_load_2d_f16(
    unsigned lds_addr, const void* gaddr,
    unsigned tensor_d0, unsigned tensor_d1,      // remaining elems per dim
    unsigned tile_d0, unsigned tile_d1,          // tile elems per dim
    unsigned long long stride0_elems)            // tensor_dim0_stride
{
    const unsigned long long ga = (unsigned long long)gaddr;
    u32x4 g0;
    g0.x = 1u;                                          // count=1, user mode
    g0.y = lds_addr;                                    // lds_addr[31:0]
    g0.z = (unsigned)(ga & 0xFFFFFFFFull);              // global_addr[31:0]
    g0.w = (unsigned)((ga >> 32) & 0x1FFFFFFull)        // global_addr[56:32]
         | (2u << 30);                                  // type = 2 (image)
    i32x8 g1;
    g1[0] = (int)((1u << 16)        // data_size = 1 -> 2 bytes
                | (1u << 20)        // pad_enable
                | (3u << 22)        // pad_interval: 16 DWORDs (64B = 32 halfs)
                | (3u << 25));      // pad_amount: 4 DWORDs (8 halfs)
    g1[1] = (int)((tensor_d0 & 0xFFFFu) << 16);                 // td0[15:0]
    g1[2] = (int)((tensor_d0 >> 16) | ((tensor_d1 & 0xFFFFu) << 16));
    g1[3] = (int)((tensor_d1 >> 16) | ((tile_d0 & 0xFFFFu) << 16));
    g1[4] = (int)(tile_d1 & 0xFFFFu);                           // tile_dim2=0
    g1[5] = (int)(unsigned)(stride0_elems & 0xFFFFFFFFull);
    g1[6] = (int)(unsigned)((stride0_elems >> 32) & 0xFFFFull); // dim1_stride=0
    g1[7] = 0;
    const i32x4 gz = {0, 0, 0, 0};
#if defined(__clang_major__) && (__clang_major__ >= 23)
    const i32x8 gz8 = {0, 0, 0, 0, 0, 0, 0, 0};
    __builtin_amdgcn_tensor_load_to_lds(g0, g1, gz, gz, gz8, 0);
#else
    __builtin_amdgcn_tensor_load_to_lds(g0, g1, gz, gz, 0);
#endif
}
#endif

// ---------------------------------------------------------------------------
// Elementwise kernels
// ---------------------------------------------------------------------------
__global__ void f32_to_f16_kernel(const float* __restrict__ in,
                                  f16* __restrict__ out, size_t n) {
    size_t i = (size_t)blockIdx.x * blockDim.x + threadIdx.x;
    size_t stride = (size_t)gridDim.x * blockDim.x;
    for (; i < n; i += stride) out[i] = (f16)in[i];
}

// x = tok_emb[seq] * sqrt(E) + pos_emb ; emits f32 + f16 copies. Block per row.
__global__ __launch_bounds__(256) void embed_kernel(
    const int* __restrict__ seq, const float* __restrict__ tok,
    const float* __restrict__ pos, float* __restrict__ x32,
    f16* __restrict__ x16) {
    const int row = blockIdx.x;           // b*S + s
    const int s = row & 255;
    const int t = seq[row];
    const int tid = threadIdx.x;
    for (int i = 0; i < 4; i++) {
        int c = i * 256 + tid;
        float v = tok[(size_t)t * 1024 + c] * 32.0f + pos[(size_t)s * 1024 + c];
        x32[(size_t)row * 1024 + c] = v;
        x16[(size_t)row * 1024 + c] = (f16)v;
    }
}

// y = xin (+ addv); out = LN(y)*g + b ; writes f32 and/or f16. Block per row.
__global__ __launch_bounds__(256) void ln_kernel(
    const float* __restrict__ xin, const float* __restrict__ addv,
    const float* __restrict__ g, const float* __restrict__ bt,
    float* __restrict__ out32, f16* __restrict__ out16) {
    __shared__ float red[256];
    const int row = blockIdx.x, tid = threadIdx.x;
    const float* xr = xin + (size_t)row * 1024;
    float y[4];
    float s = 0.f;
    for (int i = 0; i < 4; i++) {
        int c = i * 256 + tid;
        float v = xr[c];
        if (addv) v += addv[(size_t)row * 1024 + c];
        y[i] = v;
        s += v;
    }
    red[tid] = s; __syncthreads();
    for (int off = 128; off > 0; off >>= 1) {
        if (tid < off) red[tid] += red[tid + off];
        __syncthreads();
    }
    const float mean = red[0] * (1.0f / 1024.0f);
    __syncthreads();
    float sq = 0.f;
    for (int i = 0; i < 4; i++) { float d = y[i] - mean; sq += d * d; }
    red[tid] = sq; __syncthreads();
    for (int off = 128; off > 0; off >>= 1) {
        if (tid < off) red[tid] += red[tid + off];
        __syncthreads();
    }
    const float var = red[0] * (1.0f / 1024.0f);
    const float rstd = rsqrtf(var + 1e-5f);
    for (int i = 0; i < 4; i++) {
        int c = i * 256 + tid;
        float v = (y[i] - mean) * rstd * g[c] + bt[c];
        if (out32) out32[(size_t)row * 1024 + c] = v;
        if (out16) out16[(size_t)row * 1024 + c] = (f16)v;
    }
}

// ---------------------------------------------------------------------------
// Tiled WMMA GEMM:  C[M,N] = A[M,K] * W[N,K]^T + bias[N]
// BM=BN=128, BK=32. 8 waves, each wave owns a 32x64 tile (2x4 WMMA tiles).
// A tile staged by the vector path (+prefetch of next K-slab); B (weight)
// tile is workgroup-uniform -> staged by the Tensor Data Mover from wave 0,
// published with s_wait_tensorcnt + barrier.
// MODE 0: f16 out | MODE 1: f32 out | MODE 2: f16 out with exact GELU.
// ---------------------------------------------------------------------------
template <int MODE>
__global__ __launch_bounds__(256) void gemm_kernel(
    const f16* __restrict__ A, const f16* __restrict__ W,
    const float* __restrict__ bias, float* __restrict__ outf,
    f16* __restrict__ outh, int M, int N, int K) {
    __shared__ f16 As[128 * 40];   // pad 32->40 halfs: conflict-free frag reads
    __shared__ f16 Bs[128 * 40];
    const int tid = threadIdx.x;
    const int lane = tid & 31, wave = tid >> 5;
    const int wm = wave & 3, wn = wave >> 2;
    const int bm0 = blockIdx.y * 128, bn0 = blockIdx.x * 128;
    const int arow = tid >> 1, acol = (tid & 1) * 16;

    v8f acc[2][4];
    for (int i = 0; i < 2; i++)
        for (int j = 0; j < 4; j++) acc[i][j] = zero8();

    for (int kt = 0; kt < K; kt += 32) {
        __syncthreads();   // previous iteration's LDS reads are done
        {   // stage A tile: 128 x 32 halfs, 32B per thread (vector path)
            const f16* src = A + (size_t)(bm0 + arow) * K + kt + acol;
            u32x4 q0 = *(const u32x4*)src;
            u32x4 q1 = *(const u32x4*)(src + 8);
            f16* dst = As + arow * 40 + acol;
            *(u32x4*)dst = q0; *(u32x4*)(dst + 8) = q1;
            if (kt + 32 < K) __builtin_prefetch(src + 32, 0, 1);
        }
#if HAVE_TDM
        if (wave == 0) {   // stage B tile: one TDM descriptor for the block
            tdm_load_2d_f16((unsigned)(unsigned long long)(uintptr_t)Bs,
                            W + (size_t)bn0 * K + kt,
                            (unsigned)(K - kt), (unsigned)(N - bn0),
                            32u, 128u, (unsigned long long)K);
            __builtin_amdgcn_s_wait_tensorcnt(0);
        }
#else
        {   // fallback: vector staging of B tile with OOB zero-fill
            const int nrow = bn0 + arow;
            u32x4 q0 = {0u, 0u, 0u, 0u}, q1 = {0u, 0u, 0u, 0u};
            if (nrow < N) {
                const f16* src = W + (size_t)nrow * K + kt + acol;
                q0 = *(const u32x4*)src; q1 = *(const u32x4*)(src + 8);
            }
            f16* dst = Bs + arow * 40 + acol;
            *(u32x4*)dst = q0; *(u32x4*)(dst + 8) = q1;
        }
#endif
        __syncthreads();   // A ds-stores + TDM B tile visible to all waves
        v16h af[2], bf[4];
        for (int i = 0; i < 2; i++)
            af[i] = ldfrag_a(As + (wm * 32 + i * 16) * 40, 40, lane);
        for (int j = 0; j < 4; j++)
            bf[j] = ldfrag_b(Bs + (wn * 64 + j * 16) * 40, 40, lane);
        for (int i = 0; i < 2; i++)
            for (int j = 0; j < 4; j++)
                acc[i][j] = wmma_f16(af[i], bf[j], acc[i][j]);
    }

    const int r = lane & 15, hi = lane >> 4;
    for (int i = 0; i < 2; i++) {
        for (int j = 0; j < 4; j++) {
            const int col = bn0 + wn * 64 + j * 16 + r;
            if (col >= N) continue;
            const float bv = bias ? bias[col] : 0.0f;
            for (int e = 0; e < 8; e++) {
                const int row = bm0 + wm * 32 + i * 16 + e + 8 * hi;
                float v = acc[i][j][e] + bv;
                if (MODE == 2)
                    v = 0.5f * v * (1.0f + erff(v * 0.70710678118654752f));
                if (MODE == 1) outf[(size_t)row * N + col] = v;
                else           outh[(size_t)row * N + col] = (f16)v;
            }
        }
    }
}

// ---------------------------------------------------------------------------
// Attention: one block per (b, h, 64-query tile). S=256, DH=64.
// Phase 1: V^T -> LDS. Phase 2: scores = QK^T*scale + bias (WMMA; all 8 key
// fragments loaded up front so the loads clause together). Phase 3: row
// softmax. Phase 4: O = P @ V^T (WMMA) -> [B,S,E] f16.
// ---------------------------------------------------------------------------
#define SC_STRIDE 257           // f32, bank-conflict-free column scans
#define P_STRIDE  264           // f16, keeps 16B alignment for frag loads
#define SMEM_SC   (64 * SC_STRIDE * 4)     // 65792
#define SMEM_P    (64 * P_STRIDE * 2)      // 33792
#define SMEM_ATTN (SMEM_SC + 2 * SMEM_P)   // 133376 bytes

__global__ __launch_bounds__(256) void attn_kernel(
    const f16* __restrict__ qkv, const int* __restrict__ seq,
    const int* __restrict__ dist, const int* __restrict__ iso,
    const float* __restrict__ demb, const float* __restrict__ ioemb,
    const int* __restrict__ padp, f16* __restrict__ outp) {
    extern __shared__ char smem[];
    float* Sc = (float*)smem;                      // [64][257] f32 scores
    f16*   Ps = (f16*)(smem + SMEM_SC);            // [64][264] f16 probs
    f16*   Vt = (f16*)(smem + SMEM_SC + SMEM_P);   // [64][264] f16 V^T (d,k)

    const int tid = threadIdx.x, lane = tid & 31, wave = tid >> 5;
    int blk = blockIdx.x;
    const int qt = blk & 3;  blk >>= 2;
    const int h  = blk & 15; blk >>= 4;
    const int b  = blk;
    const int q0 = qt * 64;
    const f16* qbase = qkv + ((size_t)b * 256) * 3072;

    // ---- Phase 1: transpose V head-slice into LDS: Vt[d][k]
    {
        const f16* vp = qbase + (size_t)tid * 3072 + 2048 + h * 64;
        for (int d0 = 0; d0 < 64; d0 += 8) {
            H8 u; u.q = *(const u32x4*)(vp + d0);
            for (int e = 0; e < 8; e++) Vt[(d0 + e) * P_STRIDE + tid] = u.h[e];
        }
    }

    // ---- Phase 2: scores. Wave -> m-tile (16 q rows) x n-half (128 keys).
    const int mrow = (wave & 3) * 16;
    const int nh   = wave >> 2;
    const int n0   = nh * 128;
    v8f sc[8];
    for (int j = 0; j < 8; j++) sc[j] = zero8();
    for (int kk = 0; kk < 64; kk += 32) {
        v16h bfr[8];
        for (int j = 0; j < 8; j++)
            bfr[j] = ldfrag_b(
                qbase + (size_t)(n0 + j * 16) * 3072 + 1024 + h * 64 + kk,
                3072, lane);
        v16h a = ldfrag_a(qbase + (size_t)(q0 + mrow) * 3072 + h * 64 + kk,
                          3072, lane);
        for (int j = 0; j < 8; j++)
            sc[j] = wmma_f16(a, bfr[j], sc[j]);
    }
    const int pad = padp[0];
    const int r = lane & 15, hi = lane >> 4;
    for (int j = 0; j < 8; j++) {
        const int kcol = n0 + j * 16 + r;
        const bool kpad = (seq[b * 256 + kcol] == pad);
        for (int e = 0; e < 8; e++) {
            const int lq = mrow + e + 8 * hi;
            const int gq = q0 + lq;
            float sval;
            if (kcol > gq || kpad) {
                sval = -INFINITY;
            } else {
                const size_t off = ((size_t)b * 256 + gq) * 256 + kcol;
                sval = sc[j][e] * 0.125f
                     + demb[dist[off] * 16 + h] + ioemb[iso[off] * 16 + h];
            }
            Sc[lq * SC_STRIDE + kcol] = sval;
        }
    }
    __syncthreads();

    // ---- Phase 3: row-wise softmax in f32, probs -> f16
    if (tid < 64) {
        float mx = -INFINITY;
        for (int k = 0; k < 256; k++)
            mx = fmaxf(mx, Sc[tid * SC_STRIDE + k]);
        float sum = 0.f;
        for (int k = 0; k < 256; k++) {
            const float sv = Sc[tid * SC_STRIDE + k];
            const float pv = (sv > -1e37f) ? expf(sv - mx) : 0.f;
            sum += pv;
            Ps[tid * P_STRIDE + k] = (f16)pv;
        }
        const float inv = (sum > 0.f) ? 1.f / sum : 0.f;
        for (int k = 0; k < 256; k++)
            Ps[tid * P_STRIDE + k] = (f16)((float)Ps[tid * P_STRIDE + k] * inv);
    }
    __syncthreads();

    // ---- Phase 4: O = P @ V^T. Wave -> same m-tile x 32-wide d-half.
    v8f o[2]; o[0] = zero8(); o[1] = zero8();
    for (int kk = 0; kk < 256; kk += 32) {
        v16h a = ldfrag_a(Ps + mrow * P_STRIDE + kk, P_STRIDE, lane);
        for (int t = 0; t < 2; t++) {
            v16h bf = ldfrag_b(Vt + (nh * 32 + t * 16) * P_STRIDE + kk,
                               P_STRIDE, lane);
            o[t] = wmma_f16(a, bf, o[t]);
        }
    }
    for (int t = 0; t < 2; t++) {
        const int d = nh * 32 + t * 16 + r;
        for (int e = 0; e < 8; e++) {
            const int lq = mrow + e + 8 * hi;
            outp[((size_t)b * 256 + q0 + lq) * 1024 + h * 64 + d] =
                (f16)o[t][e];
        }
    }
}

// ---------------------------------------------------------------------------
// Host launcher
// ---------------------------------------------------------------------------
extern "C" void kernel_launch(void* const* d_in, const int* in_sizes, int n_in,
                              void* d_out, int out_size, void* d_ws,
                              size_t ws_size, hipStream_t stream) {
    (void)in_sizes; (void)n_in; (void)out_size; (void)ws_size;

    const int*   seq   = (const int*)  d_in[0];
    const int*   dist  = (const int*)  d_in[1];
    const int*   iso   = (const int*)  d_in[2];
    const int*   padp  = (const int*)  d_in[3];
    const float* tok   = (const float*)d_in[4];
    const float* pos   = (const float*)d_in[5];
    const float* demb  = (const float*)d_in[6];
    const float* ioemb = (const float*)d_in[7];
    const float* Wqkv  = (const float*)d_in[8];
    const float* bqkv  = (const float*)d_in[9];
    const float* Wo    = (const float*)d_in[10];
    const float* bo    = (const float*)d_in[11];
    const float* W1    = (const float*)d_in[12];
    const float* b1    = (const float*)d_in[13];
    const float* W2    = (const float*)d_in[14];
    const float* b2    = (const float*)d_in[15];
    const float* ln1s  = (const float*)d_in[16];
    const float* ln1b  = (const float*)d_in[17];
    const float* ln2s  = (const float*)d_in[18];
    const float* ln2b  = (const float*)d_in[19];
    const float* lnfs  = (const float*)d_in[20];
    const float* lnfb  = (const float*)d_in[21];
    const float* Wg    = (const float*)d_in[22];
    const float* bg    = (const float*)d_in[23];

    // workspace partition (256B aligned slabs)
    char* p = (char*)d_ws;
    auto take = [&](size_t bytes) {
        char* r = p;
        p += (bytes + 255) & ~(size_t)255;
        return r;
    };
    const size_t M = 8192;  // B*S
    f16*   wqkv16 = (f16*)  take((size_t)6 * 3072 * 1024 * 2);
    f16*   wo16   = (f16*)  take((size_t)6 * 1024 * 1024 * 2);
    f16*   w116   = (f16*)  take((size_t)6 * 4096 * 1024 * 2);
    f16*   w216   = (f16*)  take((size_t)6 * 4096 * 1024 * 2);
    f16*   wg16   = (f16*)  take((size_t)192 * 1024 * 2);
    float* x32    = (float*)take(M * 1024 * 4);
    f16*   x16    = (f16*)  take(M * 1024 * 2);
    f16*   qkv16  = (f16*)  take(M * 3072 * 2);
    f16*   ao16   = (f16*)  take(M * 1024 * 2);
    float* tmp32  = (float*)take(M * 1024 * 4);
    f16*   h16    = (f16*)  take(M * 4096 * 2);

    // weight f32 -> f16 (re-done every call: deterministic, capture-safe)
    auto cvt = [&](const float* s, f16* d, size_t n) {
        f32_to_f16_kernel<<<2048, 256, 0, stream>>>(s, d, n);
    };
    cvt(Wqkv, wqkv16, (size_t)6 * 3072 * 1024);
    cvt(Wo,   wo16,   (size_t)6 * 1024 * 1024);
    cvt(W1,   w116,   (size_t)6 * 4096 * 1024);
    cvt(W2,   w216,   (size_t)6 * 4096 * 1024);
    cvt(Wg,   wg16,   (size_t)192 * 1024);

    embed_kernel<<<M, 256, 0, stream>>>(seq, tok, pos, x32, x16);

    for (int l = 0; l < 6; l++) {
        // qkv = x @ Wqkv^T + bqkv            (f16 out)
        gemm_kernel<0><<<dim3(24, 64), 256, 0, stream>>>(
            x16, wqkv16 + (size_t)l * 3072 * 1024, bqkv + (size_t)l * 3072,
            nullptr, qkv16, (int)M, 3072, 1024);
        // attention -> ao16 [B,S,E]
        attn_kernel<<<32 * 16 * 4, 256, SMEM_ATTN, stream>>>(
            qkv16, seq, dist, iso, demb, ioemb, padp, ao16);
        // proj: tmp = ao @ Wo^T + bo         (f32 out)
        gemm_kernel<1><<<dim3(8, 64), 256, 0, stream>>>(
            ao16, wo16 + (size_t)l * 1024 * 1024, bo + (size_t)l * 1024,
            tmp32, nullptr, (int)M, 1024, 1024);
        // x = LN(x + tmp)
        ln_kernel<<<M, 256, 0, stream>>>(x32, tmp32, ln1s + (size_t)l * 1024,
                                         ln1b + (size_t)l * 1024, x32, x16);
        // h = gelu(x @ W1^T + b1)            (f16 out)
        gemm_kernel<2><<<dim3(32, 64), 256, 0, stream>>>(
            x16, w116 + (size_t)l * 4096 * 1024, b1 + (size_t)l * 4096,
            nullptr, h16, (int)M, 4096, 1024);
        // tmp = h @ W2^T + b2                (f32 out)
        gemm_kernel<1><<<dim3(8, 64), 256, 0, stream>>>(
            h16, w216 + (size_t)l * 1024 * 4096, b2 + (size_t)l * 1024,
            tmp32, nullptr, (int)M, 1024, 4096);
        // x = LN(x + tmp)
        ln_kernel<<<M, 256, 0, stream>>>(x32, tmp32, ln2s + (size_t)l * 1024,
                                         ln2b + (size_t)l * 1024, x32, x16);
    }

    // final LN -> x16, logits = x @ Wg^T + bg -> d_out (f32, N=192)
    ln_kernel<<<M, 256, 0, stream>>>(x32, nullptr, lnfs, lnfb, nullptr, x16);
    gemm_kernel<1><<<dim3(2, 64), 256, 0, stream>>>(
        x16, wg16, bg, (float*)d_out, nullptr, (int)M, 192, 1024);
}
